// Gemma3VisionEncoder_24945170055846
// MI455X (gfx1250) — compile-verified, matrix operand order checked by reading, not run on hardware
//
#include <hip/hip_runtime.h>
#include <math.h>

typedef __bf16 bf16;
typedef __attribute__((ext_vector_type(8)))  __bf16 v8bf;
typedef __attribute__((ext_vector_type(16))) __bf16 v16bf;
typedef __attribute__((ext_vector_type(8)))  float  v8f;

#define D_   1152
#define I_   4304
#define IP_  4320          /* I padded to multiple of 32 for WMMA K-steps */
#define L_   4
#define NT_  1024
#define BT_  4
#define M_   (BT_*NT_)     /* 4096 token rows */
#define H_   16
#define DH_  72
#define DHP_ 96
#define BH_  (BT_*H_)      /* 64 (batch,head) pairs */
#define PSZ_ 14
#define IMG_ 448

union FragAB { v16bf v; v8bf h[2]; };

struct EpiArgs {
  const float* bias;   // may be null
  float scale;
  bf16* out_b;         // bf16 destination
  float* out_f;        // f32 residual destination
  long strideOB;       // batch stride of out_b (elements)
  long ldo;            // row pitch of output (elements)
};

enum { EPI_QKPAD = 0, EPI_VT = 1, EPI_SCORES = 2, EPI_ATTN = 3, EPI_RESID = 4, EPI_GELU = 5 };

// ---------------------------------------------------------------------------
// Generic bf16 WMMA GEMM: C[M,N] = A[M,K] * Bt[N,K]^T (+ fused epilogue)
// A row-major [M,K], B supplied TRANSPOSED row-major [N,K] so both fragment
// loads are contiguous 16B per lane. Tile 128x128x32, 8 waves, wave = 64x32.
// Full tiles: double-buffered LDS filled by GLOBAL_LOAD_ASYNC_TO_LDS_B128
// (ASYNCcnt) overlapped with WMMA; edge tiles: synchronous staged path.
// ---------------------------------------------------------------------------
template <int EPI>
__global__ __launch_bounds__(256)
void gemm_bf16_wmma(const bf16* __restrict__ Abase, long lda, long strideA,
                    const bf16* __restrict__ Bbase, long ldb, long strideB,
                    int M, int N, int K, EpiArgs ep)
{
  __shared__ bf16 ls[2][2][128 * 40];   // [buf][A/B][row*40+col] pitch 40 -> no bank conflicts

  const int tid  = threadIdx.x;
  const int lane = tid & 31;
  const int wave = tid >> 5;
  const int wm   = wave >> 2;          // 0..1  (row half of tile)
  const int wn   = wave & 3;           // 0..3  (col quarter of tile)
  const int bz   = blockIdx.z;

  const bf16* A = Abase + (long)bz * strideA;
  const bf16* B = Bbase + (long)bz * strideB;
  const int tile_m = blockIdx.y * 128;
  const int tile_n = blockIdx.x * 128;

  v8f acc[4][2];
  #pragma unroll
  for (int i = 0; i < 4; ++i)
    #pragma unroll
    for (int j = 0; j < 2; ++j)
      #pragma unroll
      for (int e = 0; e < 8; ++e) acc[i][j][e] = 0.0f;

  const int r     = lane & 15;
  const int khalf = (lane >> 4) * 8;

  auto compute_step = [&](const bf16* baA, const bf16* baB) {
    FragAB afr[4], bfr[2];
    #pragma unroll
    for (int i = 0; i < 4; ++i) {
      const bf16* p = baA + (wm * 64 + i * 16 + r) * 40 + khalf;
      afr[i].h[0] = *(const v8bf*)p;
      afr[i].h[1] = *(const v8bf*)(p + 16);
    }
    #pragma unroll
    for (int j = 0; j < 2; ++j) {
      const bf16* p = baB + (wn * 32 + j * 16 + r) * 40 + khalf;
      bfr[j].h[0] = *(const v8bf*)p;
      bfr[j].h[1] = *(const v8bf*)(p + 16);
    }
    #pragma unroll
    for (int i = 0; i < 4; ++i)
      #pragma unroll
      for (int j = 0; j < 2; ++j)
        acc[i][j] = __builtin_amdgcn_wmma_f32_16x16x32_bf16(
            false, afr[i].v, false, bfr[j].v, (short)0, acc[i][j], false, false);
  };

  const bool full = (tile_m + 128 <= M) && (tile_n + 128 <= N) && ((K & 31) == 0);

  if (full) {
    // ---- async double-buffered pipeline ----
    unsigned long long g[4];   // per-thread global addresses (4x 16B chunks/step)
    unsigned loff[2][4];       // per-thread LDS byte addresses, both buffers
    #pragma unroll
    for (int i = 0; i < 4; ++i) {
      const int c    = tid + 256 * i;
      const int isB  = (c >= 512) ? 1 : 0;
      const int cc   = c & 511;
      const int row  = cc >> 2;
      const int col8 = (cc & 3) * 8;
      const bf16* src = isB ? B : A;
      const long  ld  = isB ? ldb : lda;
      const int   tb  = isB ? tile_n : tile_m;
      g[i] = (unsigned long long)(size_t)(src + (long)(tb + row) * ld + col8);
      loff[0][i] = (unsigned)(size_t)&ls[0][isB][row * 40 + col8];
      loff[1][i] = (unsigned)(size_t)&ls[1][isB][row * 40 + col8];
    }
    auto issue = [&](int buf) {
      #pragma unroll
      for (int i = 0; i < 4; ++i) {
        asm volatile("global_load_async_to_lds_b128 %0, %1, off"
                     :: "v"(loff[buf][i]), "v"(g[i]) : "memory");
        g[i] += 64;   // advance 32 bf16 columns
      }
    };
    issue(0);
    asm volatile("s_wait_asynccnt 0x0" ::: "memory");
    __syncthreads();
    int buf = 0;
    for (int k0 = 0; k0 < K; k0 += 32) {
      if (k0 + 32 < K) issue(buf ^ 1);        // prefetch next tile while computing
      compute_step(ls[buf][0], ls[buf][1]);
      asm volatile("s_wait_asynccnt 0x0" ::: "memory");
      __syncthreads();
      buf ^= 1;
    }
  } else {
    // ---- synchronous edge-tile path (bounds-checked, zero-filled) ----
    for (int k0 = 0; k0 < K; k0 += 32) {
      for (int c = tid; c < 1024; c += 256) {
        const bool isB  = c >= 512;
        const int  cc   = c & 511;
        const int  row  = cc >> 2;
        const int  col8 = (cc & 3) * 8;
        const bf16* src = isB ? B : A;
        const long  ld  = isB ? ldb : lda;
        const int   rb  = isB ? N : M;
        const int   gr  = (isB ? tile_n : tile_m) + row;
        const int   gk  = k0 + col8;
        v8bf val;
        if (gr < rb && gk + 8 <= K) {
          val = *(const v8bf*)(src + (long)gr * ld + gk);
        } else {
          #pragma unroll
          for (int j = 0; j < 8; ++j) {
            float f = 0.0f;
            if (gr < rb && gk + j < K) f = (float)src[(long)gr * ld + gk + j];
            val[j] = (__bf16)f;
          }
        }
        *(v8bf*)(&ls[0][isB ? 1 : 0][row * 40 + col8]) = val;
      }
      __syncthreads();
      compute_step(ls[0][0], ls[0][1]);
      __syncthreads();
    }
  }

  // epilogue: C lane layout: n = lane&15 ; m = vgpr + (lane>=16 ? 8 : 0)
  const int mh = (lane >> 4) * 8;
  #pragma unroll
  for (int i = 0; i < 4; ++i)
    #pragma unroll
    for (int j = 0; j < 2; ++j)
      #pragma unroll
      for (int e = 0; e < 8; ++e) {
        const int m = tile_m + wm * 64 + i * 16 + mh + e;
        const int n = tile_n + wn * 32 + j * 16 + r;
        if (m >= M || n >= N) continue;
        float v = acc[i][j][e];
        if (EPI == EPI_QKPAD) {
          v += ep.bias[n];
          const int bt = m >> 10, tok = m & 1023;
          const int head = n / DH_, d = n - head * DH_;
          ep.out_b[(((long)(bt * H_ + head)) * NT_ + tok) * DHP_ + d] = (bf16)v;
        } else if (EPI == EPI_VT) {
          v += ep.bias[n];
          const int bt = m >> 10, tok = m & 1023;
          const int head = n / DH_, d = n - head * DH_;
          ep.out_b[(((long)(bt * H_ + head)) * DHP_ + d) * NT_ + tok] = (bf16)v;
        } else if (EPI == EPI_SCORES) {
          ep.out_b[(long)bz * ep.strideOB + (long)m * NT_ + n] = (bf16)(v * ep.scale);
        } else if (EPI == EPI_ATTN) {
          if (n < DH_) {
            const int bt = bz >> 4, head = bz & 15;
            ep.out_b[((long)(bt * NT_ + m)) * D_ + head * DH_ + n] = (bf16)v;
          }
        } else if (EPI == EPI_RESID) {
          v += ep.bias[n];
          ep.out_f[(long)m * ep.ldo + n] += v;
        } else { // EPI_GELU (tanh approximation)
          v += ep.bias[n];
          const float u = 0.7978845608028654f * (v + 0.044715f * v * v * v);
          ep.out_b[(long)m * ep.ldo + n] = (bf16)(0.5f * v * (1.0f + tanhf(u)));
        }
      }
}

// ---------------------------------------------------------------------------
// fp32 [batch][K][N] -> bf16 [batch][N][Kp] (weight transpose+convert, padded)
// ---------------------------------------------------------------------------
__global__ __launch_bounds__(256)
void transpose_to_bf16(const float* __restrict__ src, bf16* __restrict__ dst,
                       int K, int N, long Kp, long total)
{
  const long id = (long)blockIdx.x * 256 + threadIdx.x;
  if (id >= total) return;
  const long n = id % N;
  const long k = (id / N) % K;
  const long b = id / ((long)N * K);
  dst[b * (long)N * Kp + n * Kp + k] = (bf16)src[id];
}

// ---------------------------------------------------------------------------
// Patch embedding conv (stride=kernel=14) + bias + pos_emb -> x f32 [M,D]
// ---------------------------------------------------------------------------
__global__ __launch_bounds__(256)
void patch_embed_kernel(const float* __restrict__ img, const float* __restrict__ w,
                        const float* __restrict__ cb, const float* __restrict__ pos,
                        float* __restrict__ x)
{
  const long id = (long)blockIdx.x * 256 + threadIdx.x;
  if (id >= (long)M_ * D_) return;
  const int d   = (int)(id % D_);
  const int tok = (int)(id / D_);
  const int bt  = tok >> 10;
  const int n   = tok & 1023;
  const int py  = n >> 5, px = n & 31;
  const float* ib = img + (long)bt * IMG_ * IMG_ * 3;
  float s = 0.0f;
  for (int iy = 0; iy < PSZ_; ++iy) {
    const int row = py * PSZ_ + iy;
    const float* irow = ib + ((long)row * IMG_ + px * PSZ_) * 3;  // 42 contiguous floats
    const float* wrow = w + (long)iy * (PSZ_ * 3) * D_ + d;
    #pragma unroll 6
    for (int t = 0; t < PSZ_ * 3; ++t) s += irow[t] * wrow[(long)t * D_];
  }
  x[id] = s + cb[d] + pos[(long)n * D_ + d];
}

// ---------------------------------------------------------------------------
// LayerNorm over D=1152, one block per row. OUT_BF16=1 -> bf16, else f32.
// ---------------------------------------------------------------------------
template <int OUT_BF16>
__global__ __launch_bounds__(256)
void layernorm_kernel(const float* __restrict__ x, const float* __restrict__ g,
                      const float* __restrict__ b, bf16* __restrict__ out_b,
                      float* __restrict__ out_f)
{
  __shared__ float red[256];
  const int row = blockIdx.x;
  const float* xr = x + (long)row * D_;
  float s = 0.0f, sq = 0.0f;
  for (int i = threadIdx.x; i < D_; i += 256) { const float v = xr[i]; s += v; sq += v * v; }
  red[threadIdx.x] = s; __syncthreads();
  for (int st = 128; st > 0; st >>= 1) {
    if (threadIdx.x < st) red[threadIdx.x] += red[threadIdx.x + st];
    __syncthreads();
  }
  const float mean = red[0] * (1.0f / D_);
  __syncthreads();
  red[threadIdx.x] = sq; __syncthreads();
  for (int st = 128; st > 0; st >>= 1) {
    if (threadIdx.x < st) red[threadIdx.x] += red[threadIdx.x + st];
    __syncthreads();
  }
  const float var = red[0] * (1.0f / D_) - mean * mean;
  const float rs = rsqrtf(var + 1e-6f);
  for (int i = threadIdx.x; i < D_; i += 256) {
    const float v = (xr[i] - mean) * rs * g[i] + b[i];
    if (OUT_BF16) out_b[(long)row * D_ + i] = (bf16)v;
    else          out_f[(long)row * D_ + i] = v;
  }
}

// ---------------------------------------------------------------------------
// In-place softmax over rows of 1024 bf16 scores (f32 internally)
// ---------------------------------------------------------------------------
__global__ __launch_bounds__(256)
void softmax_kernel(bf16* __restrict__ scores)
{
  __shared__ float red[256];
  bf16* p = scores + (long)blockIdx.x * NT_;
  float v[4];
  float mx = -3.0e38f;
  #pragma unroll
  for (int i = 0; i < 4; ++i) {
    v[i] = (float)p[threadIdx.x + i * 256];
    mx = fmaxf(mx, v[i]);
  }
  red[threadIdx.x] = mx; __syncthreads();
  for (int st = 128; st > 0; st >>= 1) {
    if (threadIdx.x < st) red[threadIdx.x] = fmaxf(red[threadIdx.x], red[threadIdx.x + st]);
    __syncthreads();
  }
  const float rowmax = red[0];
  __syncthreads();
  float s = 0.0f;
  #pragma unroll
  for (int i = 0; i < 4; ++i) { v[i] = __expf(v[i] - rowmax); s += v[i]; }
  red[threadIdx.x] = s; __syncthreads();
  for (int st = 128; st > 0; st >>= 1) {
    if (threadIdx.x < st) red[threadIdx.x] += red[threadIdx.x + st];
    __syncthreads();
  }
  const float inv = 1.0f / red[0];
  #pragma unroll
  for (int i = 0; i < 4; ++i) p[threadIdx.x + i * 256] = (bf16)(v[i] * inv);
}

// ---------------------------------------------------------------------------
extern "C" void kernel_launch(void* const* d_in, const int* in_sizes, int n_in,
                              void* d_out, int out_size, void* d_ws, size_t ws_size,
                              hipStream_t stream)
{
  (void)in_sizes; (void)n_in; (void)out_size; (void)ws_size;
  const float* images = (const float*)d_in[0];
  const float* conv_w = (const float*)d_in[1];
  const float* conv_b = (const float*)d_in[2];
  const float* pos    = (const float*)d_in[3];
  const float* ln1_g  = (const float*)d_in[4];
  const float* ln1_b  = (const float*)d_in[5];
  const float* Wq = (const float*)d_in[6];  const float* bq = (const float*)d_in[7];
  const float* Wk = (const float*)d_in[8];  const float* bk = (const float*)d_in[9];
  const float* Wv = (const float*)d_in[10]; const float* bv = (const float*)d_in[11];
  const float* Wo = (const float*)d_in[12]; const float* bo = (const float*)d_in[13];
  const float* ln2_g = (const float*)d_in[14]; const float* ln2_b = (const float*)d_in[15];
  const float* W1 = (const float*)d_in[16]; const float* b1 = (const float*)d_in[17];
  const float* W2 = (const float*)d_in[18]; const float* b2 = (const float*)d_in[19];
  const float* lnf_g = (const float*)d_in[20]; const float* lnf_b = (const float*)d_in[21];

  char* ws = (char*)d_ws;
  size_t off = 0;
  auto take = [&](size_t bytes) -> char* {
    char* p = ws + off;
    off = (off + bytes + 255) & ~(size_t)255;
    return p;
  };
  bf16* wq_t = (bf16*)take((size_t)L_ * D_ * D_ * 2);
  bf16* wk_t = (bf16*)take((size_t)L_ * D_ * D_ * 2);
  bf16* wv_t = (bf16*)take((size_t)L_ * D_ * D_ * 2);
  bf16* wo_t = (bf16*)take((size_t)L_ * D_ * D_ * 2);
  bf16* w1_t = (bf16*)take((size_t)L_ * (size_t)I_ * D_ * 2);   // [L][I][D]
  bf16* w2_t = (bf16*)take((size_t)L_ * (size_t)D_ * IP_ * 2);  // [L][D][IP] (K padded)
  float* x   = (float*)take((size_t)M_ * D_ * 4);
  bf16* h    = (bf16*)take((size_t)M_ * D_ * 2);
  bf16* qh   = (bf16*)take((size_t)BH_ * NT_ * DHP_ * 2);
  bf16* kh   = (bf16*)take((size_t)BH_ * NT_ * DHP_ * 2);
  bf16* vt   = (bf16*)take((size_t)BH_ * DHP_ * NT_ * 2);
  bf16* sc   = (bf16*)take((size_t)BH_ * NT_ * NT_ * 2);
  bf16* attn = (bf16*)take((size_t)M_ * D_ * 2);
  bf16* mlp1 = (bf16*)take((size_t)M_ * (size_t)IP_ * 2);       // pitch IP, pad zero

  // zero padded buffers (pads stay zero; valid lanes rewritten each layer)
  hipMemsetAsync(qh, 0, (size_t)BH_ * NT_ * DHP_ * 2, stream);
  hipMemsetAsync(kh, 0, (size_t)BH_ * NT_ * DHP_ * 2, stream);
  hipMemsetAsync(vt, 0, (size_t)BH_ * DHP_ * NT_ * 2, stream);
  hipMemsetAsync(w2_t, 0, (size_t)L_ * D_ * IP_ * 2, stream);
  hipMemsetAsync(mlp1, 0, (size_t)M_ * IP_ * 2, stream);

  // weights -> transposed bf16
  {
    const long tq = (long)L_ * D_ * D_;
    const int  gq = (int)((tq + 255) / 256);
    transpose_to_bf16<<<gq, 256, 0, stream>>>(Wq, wq_t, D_, D_, D_, tq);
    transpose_to_bf16<<<gq, 256, 0, stream>>>(Wk, wk_t, D_, D_, D_, tq);
    transpose_to_bf16<<<gq, 256, 0, stream>>>(Wv, wv_t, D_, D_, D_, tq);
    transpose_to_bf16<<<gq, 256, 0, stream>>>(Wo, wo_t, D_, D_, D_, tq);
    const long t1 = (long)L_ * D_ * I_;
    const int  g1 = (int)((t1 + 255) / 256);
    transpose_to_bf16<<<g1, 256, 0, stream>>>(W1, w1_t, D_, I_, D_, t1);   // [D][I] -> [I][D]
    transpose_to_bf16<<<g1, 256, 0, stream>>>(W2, w2_t, I_, D_, IP_, t1);  // [I][D] -> [D][IP]
  }

  // patch embedding
  {
    const long tot = (long)M_ * D_;
    patch_embed_kernel<<<(int)((tot + 255) / 256), 256, 0, stream>>>(images, conv_w, conv_b, pos, x);
  }

  const float scale = 1.0f / sqrtf((float)DH_);
  const dim3 gProj((D_ + 127) / 128, (M_ + 127) / 128, 1);      // 9 x 32
  const dim3 gMlp1((I_ + 127) / 128, (M_ + 127) / 128, 1);      // 34 x 32
  const dim3 gScore((NT_ + 127) / 128, (NT_ + 127) / 128, BH_); // 8 x 8 x 64
  const dim3 gAV((DHP_ + 127) / 128, (NT_ + 127) / 128, BH_);   // 1 x 8 x 64

  for (int l = 0; l < L_; ++l) {
    const bf16* wq_l = wq_t + (size_t)l * D_ * D_;
    const bf16* wk_l = wk_t + (size_t)l * D_ * D_;
    const bf16* wv_l = wv_t + (size_t)l * D_ * D_;
    const bf16* wo_l = wo_t + (size_t)l * D_ * D_;
    const bf16* w1_l = w1_t + (size_t)l * I_ * D_;
    const bf16* w2_l = w2_t + (size_t)l * D_ * IP_;

    // h = LN1(x) in bf16
    layernorm_kernel<1><<<M_, 256, 0, stream>>>(x, ln1_g + l * D_, ln1_b + l * D_, h, nullptr);

    EpiArgs ep{};
    // Q / K (bias + pad 72->96, per-head layout), V (bias + transposed per-head)
    ep.bias = bq + l * D_; ep.out_b = qh;
    gemm_bf16_wmma<EPI_QKPAD><<<gProj, 256, 0, stream>>>(h, D_, 0, wq_l, D_, 0, M_, D_, D_, ep);
    ep.bias = bk + l * D_; ep.out_b = kh;
    gemm_bf16_wmma<EPI_QKPAD><<<gProj, 256, 0, stream>>>(h, D_, 0, wk_l, D_, 0, M_, D_, D_, ep);
    ep.bias = bv + l * D_; ep.out_b = vt;
    gemm_bf16_wmma<EPI_VT><<<gProj, 256, 0, stream>>>(h, D_, 0, wv_l, D_, 0, M_, D_, D_, ep);

    // scores = (Q K^T) * scale, bf16, batched over 64 (bt,head)
    ep = EpiArgs{};
    ep.scale = scale; ep.out_b = sc; ep.strideOB = (long)NT_ * NT_;
    gemm_bf16_wmma<EPI_SCORES><<<gScore, 256, 0, stream>>>(
        qh, DHP_, (long)NT_ * DHP_, kh, DHP_, (long)NT_ * DHP_, NT_, NT_, DHP_, ep);

    // softmax rows (in place)
    softmax_kernel<<<BH_ * NT_, 256, 0, stream>>>(sc);

    // attn = P @ V  (heads merged back into [M, D])
    ep = EpiArgs{};
    ep.out_b = attn;
    gemm_bf16_wmma<EPI_ATTN><<<gAV, 256, 0, stream>>>(
        sc, NT_, (long)NT_ * NT_, vt, NT_, (long)DHP_ * NT_, NT_, DHP_, NT_, ep);

    // x += attn @ Wo + bo
    ep = EpiArgs{};
    ep.bias = bo + l * D_; ep.out_f = x; ep.ldo = D_;
    gemm_bf16_wmma<EPI_RESID><<<gProj, 256, 0, stream>>>(attn, D_, 0, wo_l, D_, 0, M_, D_, D_, ep);

    // h = LN2(x)
    layernorm_kernel<1><<<M_, 256, 0, stream>>>(x, ln2_g + l * D_, ln2_b + l * D_, h, nullptr);

    // mlp1 = gelu(h @ W1 + b1) in bf16 (pitch IP, pad stays zero)
    ep = EpiArgs{};
    ep.bias = b1 + l * I_; ep.out_b = mlp1; ep.ldo = IP_;
    gemm_bf16_wmma<EPI_GELU><<<gMlp1, 256, 0, stream>>>(h, D_, 0, w1_l, D_, 0, M_, I_, D_, ep);

    // x += mlp1 @ W2 + b2  (K padded to 4320 -> async fast path everywhere)
    ep = EpiArgs{};
    ep.bias = b2 + l * D_; ep.out_f = x; ep.ldo = D_;
    gemm_bf16_wmma<EPI_RESID><<<gProj, 256, 0, stream>>>(mlp1, IP_, 0, w2_l, IP_, 0, M_, D_, IP_, ep);
  }

  // final LN -> d_out (f32, [4,1024,1152] contiguous == [2,2,1024,1152])
  layernorm_kernel<0><<<M_, 256, 0, stream>>>(x, lnf_g, lnf_b, nullptr, (float*)d_out);
}
